// RNN_Decoder_61083024884284
// MI455X (gfx1250) — matrix operand order repaired and checked by reading, not compiled, not run on hardware
//
#include <hip/hip_runtime.h>
#include <hip/hip_bf16.h>

typedef unsigned short ushort_t;
typedef __attribute__((ext_vector_type(16))) __bf16 v16bf;
typedef __attribute__((ext_vector_type(8)))  __bf16 v8bf;
typedef __attribute__((ext_vector_type(8)))  float  v8f;

#define BB 32
#define SS 24
#define TT 48
#define HID 512
#define NWG 16              // 512 hidden / 32 per WG
#define TPB 512             // 16 wave32 waves
#define BUFHALF (BB*HID)    // one ping-pong half, in bf16 elements
#define XTOKSZ (32*512)     // per-WG token staging, padded to stride 512

// K-tile counts (K padded to multiples of 32)
#define KT_SC 48            // [enc 512 | prev_tok 512 | scene_h 512]
#define KT_L0 33            // [scene_h 512 | x pad32 | h0 512]
#define KT_L1 32            // [h0 512 | h1 512]

// ---------- small helpers ----------
__device__ __forceinline__ ushort_t f2bf(float f) {
    unsigned u = __builtin_bit_cast(unsigned, f);
    unsigned r = u + 0x7FFFu + ((u >> 16) & 1u);
    return (ushort_t)(r >> 16);
}
__device__ __forceinline__ float bf2f(ushort_t h) {
    unsigned u = ((unsigned)h) << 16;
    return __builtin_bit_cast(float, u);
}
__device__ __forceinline__ float sigf(float x) { return 1.f / (1.f + __expf(-x)); }

__device__ __forceinline__ v16bf cat8(v8bf lo, v8bf hi) {
    return __builtin_shufflevector(lo, hi, 0,1,2,3,4,5,6,7,8,9,10,11,12,13,14,15);
}

struct Seg {
    const ushort_t* ptr;  // bf16 row-major [32 rows][512]
    int kt_begin;         // first k-tile covered by this segment
};

#define WMMA_BF16(ALO, AHI, BLO, BHI, ACC)                                    \
    __builtin_amdgcn_wmma_f32_16x16x32_bf16(false, cat8(ALO, AHI), false,     \
                                            cat8(BLO, BHI), (short)0, (ACC),  \
                                            false, false)

#define LOADF(KIDX, A0, A1, B0, B1)                                           \
    {                                                                         \
        int _kt = (KIDX);                                                     \
        const ushort_t* _ap = aptr(_kt);                                      \
        const ushort_t* _bp = wb + (size_t)_kt * 512;                         \
        __builtin_prefetch((const void*)(_bp + 4 * 512), 0, 0);               \
        A0 = *(const v8bf*)(_ap);                                             \
        A1 = *(const v8bf*)(_ap + 16);                                        \
        B0 = *(const v8bf*)(_bp);                                             \
        B1 = *(const v8bf*)(_bp + 8);                                         \
    }

// D = A(32xK, batch-major) * B(Kx128 slice of W^T), fp32 accum, dumped to LDS gbuf.
// Wave w: mt = w&1 (batch tile), ntl = w>>1 in 0..7 -> gate g=ntl>>1, half-slice j=ntl&1.
// Flat K loop, 4-set modulo-scheduled pipeline; branchless scalar segment corrections.
template <int NSEG>
__device__ __forceinline__ void gemm_tiles(const ushort_t* __restrict__ Wp, int KT,
                                           const Seg* segs, float* gbuf, int wg) {
    const int tid  = threadIdx.x;
    const int lane = tid & 31;
    const int wv   = tid >> 5;
    const int mt   = wv & 1;
    const int ntl  = wv >> 1;
    const int g    = ntl >> 1, j = ntl & 1;
    const int ntg  = g * 32 + wg * 2 + j;     // global N-tile (of 128)
    const int half = lane >> 4;
    const int n    = lane & 15;
    const int arow = mt * 16 + n;

    const ushort_t* wb = Wp + (size_t)ntg * KT * 512 + lane * 16; // 512 ushorts/tile

    // All segments share row-stride 512, so the lane-dependent address part is
    // segment-invariant; segment switching is an additive scalar correction.
    const ushort_t* abase = segs[0].ptr + (size_t)arow * 512 + half * 8;
    const int s1b = (NSEG > 1) ? segs[1].kt_begin : 0x7fffffff;
    const int s2b = (NSEG > 2) ? segs[2].kt_begin : 0x7fffffff;
    const ptrdiff_t c1 = (NSEG > 1)
        ? (segs[1].ptr - segs[0].ptr) - (ptrdiff_t)s1b * 32 : 0;
    const ptrdiff_t c2 = (NSEG > 2)
        ? (segs[2].ptr - segs[1].ptr) - (ptrdiff_t)(s2b - s1b) * 32 : 0;

    auto aptr = [&](int kt) -> const ushort_t* {
        ptrdiff_t off = (ptrdiff_t)kt * 32;
        if (NSEG > 1) off += (kt >= s1b) ? c1 : (ptrdiff_t)0;
        if (NSEG > 2) off += (kt >= s2b) ? c2 : (ptrdiff_t)0;
        return abase + off;
    };

    v8f acc = {};
    // four fixed fragment sets (modulo schedule: slot i always holds kt % 4 == i)
    v8bf a00, a01, b00, b01;
    v8bf a10, a11, b10, b11;
    v8bf a20, a21, b20, b21;
    v8bf a30, a31, b30, b31;
    LOADF(0, a00, a01, b00, b01);     // KT >= 32 always here
    LOADF(1, a10, a11, b10, b11);
    LOADF(2, a20, a21, b20, b21);
    LOADF(3, a30, a31, b30, b31);

    int k = 0;
#pragma clang loop unroll(disable)
    for (; k + 4 < KT; k += 4) {
        acc = WMMA_BF16(a00, a01, b00, b01, acc);
        LOADF(k + 4, a00, a01, b00, b01);                          // k+4 < KT by cond
        acc = WMMA_BF16(a10, a11, b10, b11, acc);
        LOADF(min(k + 5, KT - 1), a10, a11, b10, b11);             // clamped lookahead
        acc = WMMA_BF16(a20, a21, b20, b21, acc);
        LOADF(min(k + 6, KT - 1), a20, a21, b20, b21);
        acc = WMMA_BF16(a30, a31, b30, b31, acc);
        LOADF(min(k + 7, KT - 1), a30, a31, b30, b31);
    }
    // epilogue: tiles k .. KT-1 remain in slots 0..rem-1 (uniform scalar guards)
    const int rem = KT - k;    // 1..4
    acc = WMMA_BF16(a00, a01, b00, b01, acc);
    if (rem > 1) acc = WMMA_BF16(a10, a11, b10, b11, acc);
    if (rem > 2) acc = WMMA_BF16(a20, a21, b20, b21, acc);
    if (rem > 3) acc = WMMA_BF16(a30, a31, b30, b31, acc);

    // D layout: VGPR r -> M = r + 8*half, N = lane&15
    const int growl = g * 32 + j * 16 + n;
#pragma unroll
    for (int r = 0; r < 8; ++r) {
        int batch = mt * 16 + r + half * 8;
        gbuf[batch * 128 + growl] = acc[r];
    }
}

__device__ __forceinline__ void grid_barrier(unsigned* bar, unsigned* gen) {
    __syncthreads();
    if (threadIdx.x == 0) {
        __threadfence();
        __hip_atomic_fetch_add(bar, 1u, __ATOMIC_RELEASE, __HIP_MEMORY_SCOPE_AGENT);
        unsigned target = *gen + NWG;
        while (__hip_atomic_load(bar, __ATOMIC_ACQUIRE, __HIP_MEMORY_SCOPE_AGENT) < target) {
            __builtin_amdgcn_s_sleep(1);
        }
        *gen = target;
        __threadfence();
    }
    __syncthreads();
}

// Elementwise LSTM gate update for this WG's 32 hidden units x 32 batch.
__device__ __forceinline__ void lstm_update(const float* gbuf, const float* bias,
                                            float* cstate, ushort_t* hdst,
                                            const ushort_t* hsrc_old,
                                            int eb, int eu, int hbase, bool msk) {
#pragma unroll
    for (int q = 0; q < 2; ++q) {
        int u = eu + q * 16;
        float gi = gbuf[eb * 128 + u]      + bias[hbase + u];
        float gf = gbuf[eb * 128 + 32 + u] + bias[512 + hbase + u];
        float gg = gbuf[eb * 128 + 64 + u] + bias[1024 + hbase + u];
        float go = gbuf[eb * 128 + 96 + u] + bias[1536 + hbase + u];
        float cn = sigf(gf) * cstate[q] + sigf(gi) * tanhf(gg);
        float hn = sigf(go) * tanhf(cn);
        ushort_t hv;
        if (msk) { cstate[q] = cn; hv = f2bf(hn); }
        else     { hv = hsrc_old[eb * 512 + hbase + u]; }
        hdst[eb * 512 + hbase + u] = hv;
    }
}

// ---------- prep: pack fp32 weights into bf16 WMMA-B tile format ----------
// Tile (nt,kt): 32 lanes x 8 dwords; lane's frag v = pair W[nt*16+n][kt*32 + 8*half + 16*(v>>2) + 2*(v&3)].
__global__ void k_pack_w(const float* __restrict__ Wih, const float* __restrict__ Whh,
                         ushort_t* __restrict__ dst, int KT, int c_ih, int ld_ih, int hh_start) {
    const size_t nd = (size_t)128 * KT * 256;
    unsigned* d32 = (unsigned*)dst;
    for (size_t d = (size_t)blockIdx.x * blockDim.x + threadIdx.x; d < nd;
         d += (size_t)gridDim.x * blockDim.x) {
        int v = d & 7;
        int lane = (d >> 3) & 31;
        size_t tI = d >> 8;
        int kt = (int)(tI % KT);
        int nt = (int)(tI / KT);
        int half = lane >> 4, n = lane & 15;
        int row = nt * 16 + n;
        int k0 = kt * 32 + 8 * half + 16 * (v >> 2) + 2 * (v & 3);
        float f0, f1;
        {
            int k = k0;
            f0 = (k < c_ih) ? Wih[(size_t)row * ld_ih + k]
               : (k < hh_start) ? 0.f : Whh[(size_t)row * 512 + (k - hh_start)];
            k = k0 + 1;
            f1 = (k < c_ih) ? Wih[(size_t)row * ld_ih + k]
               : (k < hh_start) ? 0.f : Whh[(size_t)row * 512 + (k - hh_start)];
        }
        d32[d] = (unsigned)f2bf(f0) | ((unsigned)f2bf(f1) << 16);
    }
}

__global__ void k_zero_convert(const float* __restrict__ enc, ushort_t* __restrict__ enc_bf,
                               ushort_t* __restrict__ zbase, size_t zcount) {
    size_t stride = (size_t)gridDim.x * blockDim.x;
    for (size_t k = (size_t)blockIdx.x * blockDim.x + threadIdx.x; k < zcount; k += stride)
        zbase[k] = 0;
    for (size_t k = (size_t)blockIdx.x * blockDim.x + threadIdx.x; k < (size_t)BB * HID; k += stride)
        enc_bf[k] = f2bf(enc[k]);
}

// ---------- persistent main kernel ----------
__global__ __launch_bounds__(TPB, 1) void rnn_persistent(
    const float* __restrict__ sketch, const int* __restrict__ lens,
    const float* __restrict__ b_sc, const float* __restrict__ b0,
    const float* __restrict__ b1, const float* __restrict__ Wout,
    const float* __restrict__ bout,
    const ushort_t* __restrict__ Wsc_p, const ushort_t* __restrict__ W0_p,
    const ushort_t* __restrict__ W1_p, const ushort_t* __restrict__ enc_bf,
    ushort_t* shbuf, ushort_t* h0buf, ushort_t* h1buf, ushort_t* xtok_all,
    unsigned* bar, float* __restrict__ out) {
    __shared__ float gbuf[32 * 128];
    const int wg  = blockIdx.x;
    const int tid = threadIdx.x;
    ushort_t* xtok = xtok_all + (size_t)wg * XTOKSZ; // per-WG tokens, stride 512
    const int eb = tid >> 4;       // batch for elementwise
    const int eu = tid & 15;       // hidden unit (and +16)
    const int hbase = wg * 32;     // this WG's hidden slice

    float csc[2] = {0.f, 0.f}, c0[2], c1[2];
    int psc = 0, p0 = 0, p1 = 0;
    unsigned barGen = 0;

    for (int s = 0; s < SS; ++s) {
        // ---- scene LSTM: X = [enc | prev_tok(=h1 final) | scene_h] ----
        {
            Seg segs[3] = {
                { enc_bf,                0 },
                { h1buf + p1 * BUFHALF,  16 },
                { shbuf + psc * BUFHALF, 32 },
            };
            gemm_tiles<3>(Wsc_p, KT_SC, segs, gbuf, wg);
        }
        __syncthreads();
        lstm_update(gbuf, b_sc, csc, shbuf + (1 - psc) * BUFHALF,
                    shbuf + psc * BUFHALF, eb, eu, hbase, true);
        grid_barrier(bar, &barGen);
        psc ^= 1;

        // ---- reset stroke state: h0=h1=c0=c1=0 ----
        c0[0] = c0[1] = c1[0] = c1[1] = 0.f;
#pragma unroll
        for (int q = 0; q < 2; ++q) {
            int u = eu + q * 16;
            h0buf[(1 - p0) * BUFHALF + eb * 512 + hbase + u] = 0;
            h1buf[(1 - p1) * BUFHALF + eb * 512 + hbase + u] = 0;
        }
        grid_barrier(bar, &barGen);
        p0 ^= 1; p1 ^= 1;

        for (int t = 0; t <= TT; ++t) {
            // token for this step (j<5; pad columns stay zero from prep)
            if (tid < 160) {
                int b = tid / 5, jj = tid % 5;
                float v;
                if (t == 0) {
                    if (s == 0) v = (jj == 2) ? 1.f : 0.f;
                    else {
                        int l = lens[b * SS + (s - 1)] - 1; if (l < 0) l = 0;
                        v = sketch[(((size_t)b * SS + (s - 1)) * TT + l) * 5 + jj];
                    }
                } else {
                    v = sketch[(((size_t)b * SS + s) * TT + (t - 1)) * 5 + jj];
                }
                xtok[b * 512 + jj] = f2bf(v);
            }
            __syncthreads();

            // ---- layer 0: X0 = [scene_h | x(pad32) | h0] ----
            {
                Seg segs[3] = {
                    { shbuf + psc * BUFHALF, 0 },
                    { xtok,                 16 },
                    { h0buf + p0 * BUFHALF, 17 },
                };
                gemm_tiles<3>(W0_p, KT_L0, segs, gbuf, wg);
            }
            __syncthreads();
            {
                int len = lens[eb * SS + s];
                bool msk = (t <= len);
                lstm_update(gbuf, b0, c0, h0buf + (1 - p0) * BUFHALF,
                            h0buf + p0 * BUFHALF, eb, eu, hbase, msk);
            }
            grid_barrier(bar, &barGen);
            p0 ^= 1;

            // ---- layer 1: X1 = [h0_new | h1] ----
            {
                Seg segs[2] = {
                    { h0buf + p0 * BUFHALF, 0 },
                    { h1buf + p1 * BUFHALF, 16 },
                };
                gemm_tiles<2>(W1_p, KT_L1, segs, gbuf, wg);
            }
            __syncthreads();
            {
                int len = lens[eb * SS + s];
                bool msk = (t <= len);
                lstm_update(gbuf, b1, c1, h1buf + (1 - p1) * BUFHALF,
                            h1buf + p1 * BUFHALF, eb, eu, hbase, msk);
            }
            grid_barrier(bar, &barGen);
            p1 ^= 1;

            // ---- output projection (torch: bias even on masked-zero rows; zero past Lmax) ----
            if (t < TT && wg == 0 && tid < 160) {
                int b = tid / 5, jj = tid % 5;
                int len = lens[b * SS + s];
                int Lmax = 0;
                for (int bb = 0; bb < 32; ++bb) {
                    int L = lens[bb * SS + s];
                    Lmax = L > Lmax ? L : Lmax;
                }
                float v = 0.f;
                if (t < Lmax) {
                    v = bout[jj];
                    if (t <= len) {
                        const ushort_t* hp = h1buf + p1 * BUFHALF + (size_t)b * 512;
                        const float* wp = Wout + jj * 512;
                        float acc = 0.f;
                        for (int k = 0; k < 512; k += 2)
                            acc += bf2f(hp[k]) * wp[k] + bf2f(hp[k + 1]) * wp[k + 1];
                        v += acc;
                    }
                }
                out[(((size_t)b * SS + s) * TT + t) * 5 + jj] = v;
            }
        } // t
    } // s
}

extern "C" void kernel_launch(void* const* d_in, const int* in_sizes, int n_in,
                              void* d_out, int out_size, void* d_ws, size_t ws_size,
                              hipStream_t stream) {
    (void)in_sizes; (void)n_in; (void)out_size; (void)ws_size;
    const float* enc     = (const float*)d_in[0];
    const float* sketch  = (const float*)d_in[1];
    const int*   lens    = (const int*)d_in[3];
    const float* Wsc_ih  = (const float*)d_in[4];
    const float* Wsc_hh  = (const float*)d_in[5];
    const float* b_sc    = (const float*)d_in[6];
    const float* W0_ih   = (const float*)d_in[7];
    const float* W0_hh   = (const float*)d_in[8];
    const float* b0      = (const float*)d_in[9];
    const float* W1_ih   = (const float*)d_in[10];
    const float* W1_hh   = (const float*)d_in[11];
    const float* b1      = (const float*)d_in[12];
    const float* Wout    = (const float*)d_in[13];
    const float* bout    = (const float*)d_in[14];
    float* out = (float*)d_out;

    char* ws = (char*)d_ws;
    unsigned* bar   = (unsigned*)ws;
    ushort_t* Wsc_p = (ushort_t*)(ws + 256);
    ushort_t* W0_p  = Wsc_p + (size_t)128 * KT_SC * 512;
    ushort_t* W1_p  = W0_p  + (size_t)128 * KT_L0 * 512;
    ushort_t* enc_bf= W1_p  + (size_t)128 * KT_L1 * 512;
    ushort_t* shbuf = enc_bf + (size_t)BB * HID;
    ushort_t* h0buf = shbuf + 2 * BUFHALF;
    ushort_t* h1buf = h0buf + 2 * BUFHALF;
    ushort_t* xtok  = h1buf + 2 * BUFHALF;

    hipMemsetAsync(bar, 0, 256, stream);
    k_pack_w<<<2048, 256, 0, stream>>>(Wsc_ih, Wsc_hh, Wsc_p, KT_SC, 1024, 1024, 1024);
    k_pack_w<<<2048, 256, 0, stream>>>(W0_ih,  W0_hh,  W0_p,  KT_L0, 517,  517,  544);
    k_pack_w<<<2048, 256, 0, stream>>>(W1_ih,  W1_hh,  W1_p,  KT_L1, 512,  512,  512);
    // zero shbuf/h0buf/h1buf/xtok (contiguous) + convert enc to bf16
    size_t zcount = (size_t)3 * 2 * BUFHALF + (size_t)NWG * XTOKSZ;
    k_zero_convert<<<256, 256, 0, stream>>>(enc, enc_bf, shbuf, zcount);

    rnn_persistent<<<NWG, TPB, 0, stream>>>(sketch, lens, b_sc, b0, b1, Wout, bout,
                                            Wsc_p, W0_p, W1_p, enc_bf,
                                            shbuf, h0buf, h1buf, xtok, bar, out);
}
